// TransformerMemoryNet_17566416241318
// MI455X (gfx1250) — matrix operand order repaired and checked by reading, not compiled, not run on hardware
//
#include <hip/hip_runtime.h>
#include <math.h>

// ---------------------------------------------------------------------------
// Problem constants (from reference)
// ---------------------------------------------------------------------------
static constexpr int E_SZ = 10000, R_SZ = 200, D_ = 64, DFF_ = 256;
static constexpr int L_ = 2, H_ = 8, B_ = 8, S_ = 1024;
static constexpr int N_ = B_ * S_;          // 8192 quadruples / tokens
static constexpr int QPQ_ = 3;
static constexpr int Q_ = N_ * QPQ_;        // 24576 qualifiers

typedef __attribute__((ext_vector_type(16))) _Float16 v16h;
typedef __attribute__((ext_vector_type(8)))  float    v8f;

// D = A*B + C, 16x16 tile, K=32, f16 in / f32 accumulate
__device__ inline v8f wmma_f16(v16h a, v16h b, v8f c) {
  return __builtin_amdgcn_wmma_f32_16x16x32_f16(
      /*neg_a=*/false, a, /*neg_b=*/false, b,
      /*c_mod=*/(short)0, c, /*reuse_a=*/false, /*reuse_b=*/false);
}

// ---------------------------------------------------------------------------
// Fragment loaders (layouts per CDNA5 ISA 7.12.2, wave32)
//   A (16x32, f16): lane holds row = lane&15; k = 16*(v>>2) + 8*(lane>>4) + 2*(v&3)
//   B (32x16, f16): lane holds col = lane&15; k = 16*(v>>2) + 8*(lane>>4) + 2*(v&3)
//   C/D (16x16 f32): lane holds col = lane&15; rows 8*(lane>>4)+r in acc[r]
// ---------------------------------------------------------------------------
__device__ inline v16h a_frag_lds(const _Float16* base, int stride, int kb, int lane) {
  const int row = lane & 15, hi = lane >> 4;
  v16h a;
#pragma unroll
  for (int v = 0; v < 8; ++v) {
    const int k0 = kb + ((v >> 2) << 4) + (hi << 3) + ((v & 3) << 1);
    const _Float16* p = base + row * stride + k0;
    a[2 * v]     = p[0];
    a[2 * v + 1] = p[1];
  }
  return a;
}

__device__ inline v16h a_frag_g(const float* base, int stride, int kb, int lane) {
  const int row = lane & 15, hi = lane >> 4;
  v16h a;
#pragma unroll
  for (int v = 0; v < 8; ++v) {
    const int k0 = kb + ((v >> 2) << 4) + (hi << 3) + ((v & 3) << 1);
    const float* p = base + row * stride + k0;
    a[2 * v]     = (_Float16)p[0];
    a[2 * v + 1] = (_Float16)p[1];
  }
  return a;
}

// Pre-packed B fragment: one aligned 32-byte load per lane.
// Packed layout: tile (kt, nt) at ((kt*ntn + nt)*32 + lane)*16 halves.
__device__ inline v16h b_frag_p(const _Float16* packed, int ntn, int kt, int nt,
                                int lane) {
  return *(const v16h*)(packed + (((size_t)(kt * ntn + nt) * 32) + lane) * 16);
}

// ===========================================================================
// Kernel 0: pack a row-major f32 weight W[K x ldn] (Nout == ldn) into
//           fragment-major f16.  One lane per fragment slot.
// ===========================================================================
__global__ void k_pack(const float* W, int ldn, int ktiles, _Float16* out) {
  const int tid = blockIdx.x * 256 + threadIdx.x;
  const int lane = tid & 31;
  const int tile = tid >> 5;
  const int ntn = ldn >> 4;
  if (tile >= ktiles * ntn) return;
  const int kt = tile / ntn, nt = tile - kt * ntn;
  const int col = lane & 15, hi = lane >> 4;
  v16h b;
#pragma unroll
  for (int v = 0; v < 8; ++v) {
    const int k = kt * 32 + ((v >> 2) << 4) + (hi << 3) + ((v & 3) << 1);
    b[2 * v]     = (_Float16)W[(size_t)k * ldn + nt * 16 + col];
    b[2 * v + 1] = (_Float16)W[(size_t)(k + 1) * ldn + nt * 16 + col];
  }
  *(v16h*)(out + ((size_t)tile * 32 + lane) * 16) = b;
}

// ===========================================================================
// Kernel 1: qualifier MLP  pq = relu([rel|ent] @ W1 + b1) @ W2 + b2  and
//           attention scores s = pq . attn_w.  8 waves/WG, 16 rows/wave.
// ===========================================================================
__global__ void k_qualmlp(const int* qt, const int* qv,
                          const float* ent, const float* rel,
                          const _Float16* W1p, const float* b1,
                          const _Float16* W2p, const float* b2,
                          const float* attw,
                          float* pq_out, float* s_out) {
  __shared__ _Float16 sh_h[8][16 * 64];   // hidden activations (f16)
  __shared__ float    sh_p[8][16 * 64];   // pq tile (f32) for score dot
  const int lane = threadIdx.x & 31, wv = threadIdx.x >> 5;
  const int row0 = (blockIdx.x * 8 + wv) * 16;
  const int row = lane & 15, hi = lane >> 4;

  const int r = row0 + row;
  const float* relrow = rel + qt[r] * D_;
  const float* entrow = ent + qv[r] * D_;

  // ---- layer 1: K = 128 (concat gathered on the fly), Nout = 64
  v8f acc[4] = {};
#pragma unroll
  for (int ks = 0; ks < 4; ++ks) {
    v16h a;
#pragma unroll
    for (int v = 0; v < 8; ++v) {
      const int k0 = ks * 32 + ((v >> 2) << 4) + (hi << 3) + ((v & 3) << 1);
      const float x0 = (k0 < 64) ? relrow[k0] : entrow[k0 - 64];
      const float x1 = (k0 + 1 < 64) ? relrow[k0 + 1] : entrow[k0 + 1 - 64];
      a[2 * v]     = (_Float16)x0;
      a[2 * v + 1] = (_Float16)x1;
    }
#pragma unroll
    for (int t = 0; t < 4; ++t)
      acc[t] = wmma_f16(a, b_frag_p(W1p, 4, ks, t, lane), acc[t]);
  }
  // bias + relu -> LDS (f16)
#pragma unroll
  for (int t = 0; t < 4; ++t) {
    const int n = t * 16 + row;
    const float bb = b1[n];
#pragma unroll
    for (int rr = 0; rr < 8; ++rr) {
      float hv = acc[t][rr] + bb;
      hv = hv > 0.f ? hv : 0.f;
      sh_h[wv][(hi * 8 + rr) * 64 + n] = (_Float16)hv;
    }
  }
  __builtin_amdgcn_wave_barrier();

  // ---- layer 2: K = 64, Nout = 64
  v8f acc2[4] = {};
#pragma unroll
  for (int ks = 0; ks < 2; ++ks) {
    v16h a = a_frag_lds(sh_h[wv], 64, ks * 32, lane);
#pragma unroll
    for (int t = 0; t < 4; ++t)
      acc2[t] = wmma_f16(a, b_frag_p(W2p, 4, ks, t, lane), acc2[t]);
  }
#pragma unroll
  for (int t = 0; t < 4; ++t) {
    const int n = t * 16 + row;
    const float bb = b2[n];
#pragma unroll
    for (int rr = 0; rr < 8; ++rr) {
      const float p = acc2[t][rr] + bb;
      const int rrow = hi * 8 + rr;
      sh_p[wv][rrow * 64 + n] = p;
      pq_out[(size_t)(row0 + rrow) * 64 + n] = p;
    }
  }
  __builtin_amdgcn_wave_barrier();
  // per-row score dot (lanes 0..15, one row each)
  if (lane < 16) {
    float sum = 0.f;
    for (int j = 0; j < 64; ++j) sum += sh_p[wv][lane * 64 + j] * attw[j];
    s_out[row0 + lane] = sum;
  }
}

// ===========================================================================
// Kernel 2: segment softmax over 3 qualifiers + token assembly + projection
//           x = [ent[h] | rel[r] | ent[t] | agg] @ proj_W + proj_b
//           4 waves/WG, 16 rows/wave, K = 256.
// ===========================================================================
__global__ void k_token(const int* head, const int* relidx, const int* tail,
                        const float* ent, const float* rel,
                        const float* pq, const float* s,
                        const _Float16* projWp, const float* projb,
                        float* x) {
  __shared__ _Float16 sh_tok[4][16 * 256];
  const int lane = threadIdx.x & 31, wv = threadIdx.x >> 5;
  const int q0 = (blockIdx.x * 4 + wv) * 16;
  const int row = lane & 15, hi = lane >> 4;
  const int qd = q0 + row;

  // softmax over the 3 qualifiers of this quadruple (replicated per half)
  const float s0 = s[3 * qd + 0], s1 = s[3 * qd + 1], s2 = s[3 * qd + 2];
  const float mm = fmaxf(s0, fmaxf(s1, s2));
  float e0 = __expf(s0 - mm), e1 = __expf(s1 - mm), e2 = __expf(s2 - mm);
  const float inv = 1.f / (e0 + e1 + e2);
  e0 *= inv; e1 *= inv; e2 *= inv;
  const float* p0 = pq + (size_t)(3 * qd) * 64;

  const float* hrow = ent + head[qd] * D_;
  const float* rrow = rel + relidx[qd] * D_;
  const float* trow = ent + tail[qd] * D_;
  // each lane fills half of its row (128 of 256 cols)
  for (int j = 0; j < 128; ++j) {
    const int c = hi * 128 + j;
    float v;
    if (c < 64)       v = hrow[c];
    else if (c < 128) v = rrow[c - 64];
    else if (c < 192) v = trow[c - 128];
    else {
      const int d = c - 192;
      v = e0 * p0[d] + e1 * p0[64 + d] + e2 * p0[128 + d];
    }
    sh_tok[wv][row * 256 + c] = (_Float16)v;
  }
  __builtin_amdgcn_wave_barrier();

  v8f acc[4] = {};
#pragma unroll
  for (int ks = 0; ks < 8; ++ks) {
    v16h a = a_frag_lds(sh_tok[wv], 256, ks * 32, lane);
#pragma unroll
    for (int t = 0; t < 4; ++t)
      acc[t] = wmma_f16(a, b_frag_p(projWp, 4, ks, t, lane), acc[t]);
  }
#pragma unroll
  for (int t = 0; t < 4; ++t) {
    const int n = t * 16 + row;
    const float bb = projb[n];
#pragma unroll
    for (int rr = 0; rr < 8; ++rr)
      x[(size_t)(q0 + hi * 8 + rr) * 64 + n] = acc[t][rr] + bb;
  }
}

// ===========================================================================
// Kernel 3: QKV projection, scatter into [B][H][S][8] per-head buffers.
// ===========================================================================
__global__ void k_qkv(const float* x, const _Float16* Wp, const float* bqkv,
                      float* qb, float* kb, float* vb) {
  const int lane = threadIdx.x & 31, wv = threadIdx.x >> 5;
  const int row0 = (blockIdx.x * 8 + wv) * 16;
  const int row = lane & 15, hi = lane >> 4;

  const v16h a0 = a_frag_g(x + (size_t)row0 * 64, 64, 0, lane);
  const v16h a1 = a_frag_g(x + (size_t)row0 * 64, 64, 32, lane);
  v8f acc[12] = {};
#pragma unroll
  for (int t = 0; t < 12; ++t) {
    acc[t] = wmma_f16(a0, b_frag_p(Wp, 12, 0, t, lane), acc[t]);
    acc[t] = wmma_f16(a1, b_frag_p(Wp, 12, 1, t, lane), acc[t]);
  }
#pragma unroll
  for (int t = 0; t < 12; ++t) {
    const int n = t * 16 + row;
    const float bb = bqkv[n];
    float* dst; int dn;
    if (n < 64)       { dst = qb; dn = n; }
    else if (n < 128) { dst = kb; dn = n - 64; }
    else              { dst = vb; dn = n - 128; }
    const int h = dn >> 3, c = dn & 7;
#pragma unroll
    for (int rr = 0; rr < 8; ++rr) {
      const int gr = row0 + hi * 8 + rr;
      const int bi = gr >> 10, si = gr & 1023;
      dst[(size_t)(((bi * H_ + h) << 10) + si) * 8 + c] = acc[t][rr] + bb;
    }
  }
}

// ===========================================================================
// Kernel 4: flash attention. One wave = one 16-query tile of one (b,h).
//           dh=8 zero-padded to K=32 for QK^T; 32-key blocks; online softmax
//           with cross-lane shuffles; P@V via one WMMA per block.
// ===========================================================================
__global__ void k_attn(const float* qb, const float* kbuf, const float* vbuf,
                       float* ob) {
  __shared__ _Float16 sh_p[8][16 * 32];
  const int lane = threadIdx.x & 31, wv = threadIdx.x >> 5;
  const int unit = blockIdx.x * 8 + wv;    // 0..B*H*(S/16)-1
  const int qt = unit & 63;                // query tile within sequence
  const int bh = unit >> 6;                // (b*H + h)
  const int row = lane & 15, hi = lane >> 4;

  const float* Qp = qb   + (size_t)(bh << 10) * 8;
  const float* Kp = kbuf + (size_t)(bh << 10) * 8;
  const float* Vp = vbuf + (size_t)(bh << 10) * 8;
  const float scale = 0.35355339059327373f;  // 1/sqrt(8)

  // Q fragment (scale folded in; k >= 8 zero-padded)
  v16h aq;
#pragma unroll
  for (int v = 0; v < 8; ++v) {
    const int k0 = ((v >> 2) << 4) + (hi << 3) + ((v & 3) << 1);
    float x0 = 0.f, x1 = 0.f;
    if (k0 < 8) {
      const float* p = Qp + (size_t)(qt * 16 + row) * 8 + k0;
      x0 = p[0] * scale; x1 = p[1] * scale;
    }
    aq[2 * v] = (_Float16)x0; aq[2 * v + 1] = (_Float16)x1;
  }

  v8f oacc = {};
  float m_r[8], l_r[8];
#pragma unroll
  for (int rr = 0; rr < 8; ++rr) { m_r[rr] = -1e30f; l_r[rr] = 0.f; }

  for (int kb2 = 0; kb2 < 64; kb2 += 2) {      // 32 keys per iteration
    v8f st[2];
#pragma unroll
    for (int half = 0; half < 2; ++half) {
      const int key0 = (kb2 + half) * 16;
      v16h bk;                                  // B[k][col] = K[key0+col][k]
#pragma unroll
      for (int v = 0; v < 8; ++v) {
        const int k = ((v >> 2) << 4) + (hi << 3) + ((v & 3) << 1);
        float x0 = 0.f, x1 = 0.f;
        if (k < 8) {
          const float* p = Kp + (size_t)(key0 + (lane & 15)) * 8 + k;
          x0 = p[0]; x1 = p[1];
        }
        bk[2 * v] = (_Float16)x0; bk[2 * v + 1] = (_Float16)x1;
      }
      v8f z = {};
      st[half] = wmma_f16(aq, bk, z);
    }
    // online softmax: row max / row sum across the 16 lanes of each half
    float mnew[8], rs[8];
#pragma unroll
    for (int rr = 0; rr < 8; ++rr) {
      float t = fmaxf(st[0][rr], st[1][rr]);
      for (int off = 1; off < 16; off <<= 1)
        t = fmaxf(t, __shfl_xor(t, off, 32));
      mnew[rr] = fmaxf(m_r[rr], t);
    }
#pragma unroll
    for (int rr = 0; rr < 8; ++rr) {
      const float p0 = __expf(st[0][rr] - mnew[rr]);
      const float p1 = __expf(st[1][rr] - mnew[rr]);
      sh_p[wv][(hi * 8 + rr) * 32 + row]      = (_Float16)p0;
      sh_p[wv][(hi * 8 + rr) * 32 + 16 + row] = (_Float16)p1;
      float t = p0 + p1;
      for (int off = 1; off < 16; off <<= 1)
        t += __shfl_xor(t, off, 32);
      rs[rr] = t;
    }
#pragma unroll
    for (int rr = 0; rr < 8; ++rr) {
      const float corr = __expf(m_r[rr] - mnew[rr]);
      l_r[rr] = l_r[rr] * corr + rs[rr];
      oacc[rr] = oacc[rr] * corr;
      m_r[rr] = mnew[rr];
    }
    __builtin_amdgcn_wave_barrier();
    // P (16x32, from LDS) @ V (32x8 padded to 16 cols)
    const v16h ap = a_frag_lds(sh_p[wv], 32, 0, lane);
    v16h bv;
    const int key0 = kb2 * 16;
    const int col = lane & 15;
#pragma unroll
    for (int v = 0; v < 8; ++v) {
      const int k = ((v >> 2) << 4) + (hi << 3) + ((v & 3) << 1);
      float x0 = 0.f, x1 = 0.f;
      if (col < 8) {
        x0 = Vp[(size_t)(key0 + k) * 8 + col];
        x1 = Vp[(size_t)(key0 + k + 1) * 8 + col];
      }
      bv[2 * v] = (_Float16)x0; bv[2 * v + 1] = (_Float16)x1;
    }
    oacc = wmma_f16(ap, bv, oacc);
  }

  const int col = lane & 15;
  if (col < 8) {
    const int h = bh & 7, bi = bh >> 3;
#pragma unroll
    for (int rr = 0; rr < 8; ++rr) {
      const int srow = qt * 16 + hi * 8 + rr;
      ob[(size_t)((bi << 10) + srow) * 64 + h * 8 + col] = oacc[rr] / l_r[rr];
    }
  }
}

// ===========================================================================
// Kernel 5: O-projection + residual + LayerNorm1 (in-place update of x)
// ===========================================================================
__global__ void k_oproj_ln(const float* ob, const _Float16* Wop, const float* bo,
                           const float* ln_s, const float* ln_b, float* x) {
  __shared__ float sh_y[8][16 * 64];
  __shared__ float sh_st[8][16][2];
  const int lane = threadIdx.x & 31, wv = threadIdx.x >> 5;
  const int row0 = (blockIdx.x * 8 + wv) * 16;
  const int row = lane & 15, hi = lane >> 4;

  const v16h a0 = a_frag_g(ob + (size_t)row0 * 64, 64, 0, lane);
  const v16h a1 = a_frag_g(ob + (size_t)row0 * 64, 64, 32, lane);
  v8f acc[4] = {};
#pragma unroll
  for (int t = 0; t < 4; ++t) {
    acc[t] = wmma_f16(a0, b_frag_p(Wop, 4, 0, t, lane), acc[t]);
    acc[t] = wmma_f16(a1, b_frag_p(Wop, 4, 1, t, lane), acc[t]);
  }
#pragma unroll
  for (int t = 0; t < 4; ++t) {
    const int n = t * 16 + row;
    const float bb = bo[n];
#pragma unroll
    for (int rr = 0; rr < 8; ++rr) {
      const int rrow = hi * 8 + rr;
      sh_y[wv][rrow * 64 + n] = x[(size_t)(row0 + rrow) * 64 + n] + acc[t][rr] + bb;
    }
  }
  __builtin_amdgcn_wave_barrier();
  if (lane < 16) {
    float mean = 0.f;
    for (int j = 0; j < 64; ++j) mean += sh_y[wv][lane * 64 + j];
    mean *= (1.f / 64.f);
    float var = 0.f;
    for (int j = 0; j < 64; ++j) {
      const float d = sh_y[wv][lane * 64 + j] - mean;
      var += d * d;
    }
    var *= (1.f / 64.f);
    sh_st[wv][lane][0] = mean;
    sh_st[wv][lane][1] = rsqrtf(var + 1e-5f);
  }
  __builtin_amdgcn_wave_barrier();
  const float mean = sh_st[wv][row][0], inv = sh_st[wv][row][1];
  for (int j = 0; j < 32; ++j) {
    const int c = hi * 32 + j;
    x[(size_t)(row0 + row) * 64 + c] =
        (sh_y[wv][row * 64 + c] - mean) * inv * ln_s[c] + ln_b[c];
  }
}

// ===========================================================================
// Kernel 6: FFN (64 -> 256 relu -> 64) + residual + LayerNorm2
// ===========================================================================
__global__ void k_ffn_ln(const float* xin, const _Float16* W1p, const float* b1,
                         const _Float16* W2p, const float* b2,
                         const float* ln_s, const float* ln_b, float* xout) {
  __shared__ _Float16 sh_h[4][16 * 256];
  __shared__ float    sh_y[4][16 * 64];
  __shared__ float    sh_st[4][16][2];
  const int lane = threadIdx.x & 31, wv = threadIdx.x >> 5;
  const int row0 = (blockIdx.x * 4 + wv) * 16;
  const int row = lane & 15, hi = lane >> 4;

  const v16h a0 = a_frag_g(xin + (size_t)row0 * 64, 64, 0, lane);
  const v16h a1 = a_frag_g(xin + (size_t)row0 * 64, 64, 32, lane);

  // GEMM1: Nout=256, computed in 4 chunks of 64 to bound VGPR pressure
  for (int ch = 0; ch < 4; ++ch) {
    v8f acc[4] = {};
#pragma unroll
    for (int t = 0; t < 4; ++t) {
      const int nt = ch * 4 + t;
      acc[t] = wmma_f16(a0, b_frag_p(W1p, 16, 0, nt, lane), acc[t]);
      acc[t] = wmma_f16(a1, b_frag_p(W1p, 16, 1, nt, lane), acc[t]);
    }
#pragma unroll
    for (int t = 0; t < 4; ++t) {
      const int n = ch * 64 + t * 16 + row;
      const float bb = b1[n];
#pragma unroll
      for (int rr = 0; rr < 8; ++rr) {
        float hv = acc[t][rr] + bb;
        hv = hv > 0.f ? hv : 0.f;
        sh_h[wv][(hi * 8 + rr) * 256 + n] = (_Float16)hv;
      }
    }
  }
  __builtin_amdgcn_wave_barrier();

  // GEMM2: K = 256
  v8f acc2[4] = {};
#pragma unroll
  for (int ks = 0; ks < 8; ++ks) {
    v16h a = a_frag_lds(sh_h[wv], 256, ks * 32, lane);
#pragma unroll
    for (int t = 0; t < 4; ++t)
      acc2[t] = wmma_f16(a, b_frag_p(W2p, 4, ks, t, lane), acc2[t]);
  }
#pragma unroll
  for (int t = 0; t < 4; ++t) {
    const int n = t * 16 + row;
    const float bb = b2[n];
#pragma unroll
    for (int rr = 0; rr < 8; ++rr) {
      const int rrow = hi * 8 + rr;
      sh_y[wv][rrow * 64 + n] =
          xin[(size_t)(row0 + rrow) * 64 + n] + acc2[t][rr] + bb;
    }
  }
  __builtin_amdgcn_wave_barrier();
  if (lane < 16) {
    float mean = 0.f;
    for (int j = 0; j < 64; ++j) mean += sh_y[wv][lane * 64 + j];
    mean *= (1.f / 64.f);
    float var = 0.f;
    for (int j = 0; j < 64; ++j) {
      const float d = sh_y[wv][lane * 64 + j] - mean;
      var += d * d;
    }
    var *= (1.f / 64.f);
    sh_st[wv][lane][0] = mean;
    sh_st[wv][lane][1] = rsqrtf(var + 1e-5f);
  }
  __builtin_amdgcn_wave_barrier();
  const float mean = sh_st[wv][row][0], inv = sh_st[wv][row][1];
  for (int j = 0; j < 32; ++j) {
    const int c = hi * 32 + j;
    xout[(size_t)(row0 + row) * 64 + c] =
        (sh_y[wv][row * 64 + c] - mean) * inv * ln_s[c] + ln_b[c];
  }
}

// ===========================================================================
// Host launcher
// ===========================================================================
extern "C" void kernel_launch(void* const* d_in, const int* in_sizes, int n_in,
                              void* d_out, int out_size, void* d_ws, size_t ws_size,
                              hipStream_t stream) {
  (void)in_sizes; (void)n_in; (void)out_size; (void)ws_size;
  const int* head_idx = (const int*)d_in[0];
  const int* rel_idx  = (const int*)d_in[1];
  const int* tail_idx = (const int*)d_in[2];
  const int* qtype    = (const int*)d_in[3];
  const int* qval     = (const int*)d_in[4];
  // d_in[5] qual_seg: structure is fixed (3 per quad, contiguous) — unused
  const float* ent  = (const float*)d_in[6];
  const float* rel  = (const float*)d_in[7];
  const float* mW1  = (const float*)d_in[8];
  const float* mb1  = (const float*)d_in[9];
  const float* mW2  = (const float*)d_in[10];
  const float* mb2  = (const float*)d_in[11];
  const float* attw = (const float*)d_in[12];
  const float* pW   = (const float*)d_in[13];
  const float* pb   = (const float*)d_in[14];
  const float* Wqkv = (const float*)d_in[15];
  const float* bqkv = (const float*)d_in[16];
  const float* Wo   = (const float*)d_in[17];
  const float* bo   = (const float*)d_in[18];
  const float* l1s  = (const float*)d_in[19];
  const float* l1b  = (const float*)d_in[20];
  const float* Wf1  = (const float*)d_in[21];
  const float* bf1  = (const float*)d_in[22];
  const float* Wf2  = (const float*)d_in[23];
  const float* bf2  = (const float*)d_in[24];
  const float* l2s  = (const float*)d_in[25];
  const float* l2b  = (const float*)d_in[26];

  // workspace layout (f32): pq[Q,64] s[Q] x[N,64] q/k/v[N,64] o[N,64]
  float* ws   = (float*)d_ws;
  float* pq   = ws;
  float* sbuf = pq   + (size_t)Q_ * 64;
  float* xbuf = sbuf + (size_t)Q_;
  float* qb   = xbuf + (size_t)N_ * 64;
  float* kb   = qb   + (size_t)N_ * 64;
  float* vb   = kb   + (size_t)N_ * 64;
  float* ob   = vb   + (size_t)N_ * 64;
  // packed f16 weight region (fragment-major); 32-byte aligned by construction
  _Float16* wpk   = (_Float16*)(ob + (size_t)N_ * 64);
  _Float16* W1p   = wpk;                                   // 128x64  -> 8192
  _Float16* W2p   = W1p  + 128 * 64;                       // 64x64   -> 4096
  _Float16* pWp   = W2p  + 64 * 64;                        // 256x64  -> 16384
  _Float16* qkvp  = pWp  + 256 * 64;                       // L*64x192-> 24576
  _Float16* wop   = qkvp + (size_t)L_ * 64 * 192;          // L*64x64 -> 8192
  _Float16* f1p   = wop  + (size_t)L_ * 64 * 64;           // L*64x256-> 32768
  _Float16* f2p   = f1p  + (size_t)L_ * 64 * 256;          // L*256x64-> 32768

  // ---- weight packing (layer axis folded into K-tiles; rows contiguous)
  // blocks = ceil(ktiles * (ldn/16) * 32 / 256)
  k_pack<<<2, 256, 0, stream>>>(mW1, 64, 4, W1p);          // K=128
  k_pack<<<1, 256, 0, stream>>>(mW2, 64, 2, W2p);          // K=64
  k_pack<<<4, 256, 0, stream>>>(pW, 64, 8, pWp);           // K=256
  k_pack<<<6, 256, 0, stream>>>(Wqkv, 192, 4, qkvp);       // K=L*64=128
  k_pack<<<2, 256, 0, stream>>>(Wo, 64, 4, wop);           // K=128
  k_pack<<<8, 256, 0, stream>>>(Wf1, 256, 4, f1p);         // K=128
  k_pack<<<8, 256, 0, stream>>>(Wf2, 64, 16, f2p);         // K=L*256=512

  // ---- tokenizer
  k_qualmlp<<<Q_ / 128, 256, 0, stream>>>(qtype, qval, ent, rel,
                                          W1p, mb1, W2p, mb2, attw, pq, sbuf);
  k_token<<<N_ / 64, 128, 0, stream>>>(head_idx, rel_idx, tail_idx, ent, rel,
                                       pq, sbuf, pWp, pb, xbuf);
  // ---- transformer encoder
  for (int i = 0; i < L_; ++i) {
    // per-layer packed slices: layer i = k-tiles [2i, 2i+1] (or [8i..8i+7] for Wff2)
    const _Float16* qkv_i = qkvp + (size_t)i * 2 * 12 * 512;
    const _Float16* wo_i  = wop  + (size_t)i * 2 * 4 * 512;
    const _Float16* f1_i  = f1p  + (size_t)i * 2 * 16 * 512;
    const _Float16* f2_i  = f2p  + (size_t)i * 8 * 4 * 512;
    k_qkv<<<N_ / 128, 256, 0, stream>>>(xbuf, qkv_i, bqkv + i * 192, qb, kb, vb);
    k_attn<<<(B_ * H_ * (S_ / 16)) / 8, 256, 0, stream>>>(qb, kb, vb, ob);
    k_oproj_ln<<<N_ / 128, 256, 0, stream>>>(ob, wo_i, bo + i * 64,
                                             l1s + i * 64, l1b + i * 64, xbuf);
    float* xout = (i == L_ - 1) ? (float*)d_out : xbuf;
    k_ffn_ln<<<N_ / 64, 128, 0, stream>>>(xbuf, f1_i, bf1 + i * 256,
                                          f2_i, bf2 + i * 64,
                                          l2s + i * 64, l2b + i * 64, xout);
  }
}
